// WATModel_30623116820693
// MI455X (gfx1250) — matrix operand order repaired and checked by reading, not compiled, not run on hardware
//
#include <hip/hip_runtime.h>
#include <hip/hip_bf16.h>
#include <math.h>

#define WB 32
#define WS 2048
#define WD 1024
#define WV 32000
#define KSTEP 32
#define BN 128            // block column tile for staged GEMMs

typedef __bf16 bf16_t;
typedef __attribute__((ext_vector_type(16))) __bf16 v16bf;
typedef __attribute__((ext_vector_type(8)))  __bf16 v8bf;
typedef __attribute__((ext_vector_type(8)))  float  v8f;

#ifndef __has_builtin
#define __has_builtin(x) 0
#endif
#if __has_builtin(__builtin_amdgcn_global_load_async_to_lds_b128)
#define ASYNC_LDS 1
#else
#define ASYNC_LDS 0
#endif

#if ASYNC_LDS
typedef int v4i_ __attribute__((vector_size(16)));
typedef __attribute__((address_space(1))) v4i_* g128p;
typedef __attribute__((address_space(3))) v4i_* l128p;
#endif

static __device__ __forceinline__ float  bf2f(bf16_t x) { return (float)x; }
static __device__ __forceinline__ bf16_t f2bf(float  x) { return (bf16_t)x; }
static __device__ __forceinline__ float  sigmoidf_(float x) { return 1.0f / (1.0f + __expf(-x)); }

static __device__ __forceinline__ v8f wmma_bf16(v16bf a, v16bf b, v8f c) {
  return __builtin_amdgcn_wmma_f32_16x16x32_bf16(false, a, false, b, (short)0, c, false, false);
}

// async (or fallback) 16B global -> LDS copy
static __device__ __forceinline__ void cp_b128(const bf16_t* g, bf16_t* l) {
#if ASYNC_LDS
  __builtin_amdgcn_global_load_async_to_lds_b128((g128p)g, (l128p)l, 0, 0);
#else
  *(v8bf*)l = *(const v8bf*)g;
#endif
}

static __device__ __forceinline__ void wait_async() {
#if ASYNC_LDS
#if __has_builtin(__builtin_amdgcn_s_wait_asynccnt)
  __builtin_amdgcn_s_wait_asynccnt(0);
#else
  asm volatile("s_wait_asynccnt 0x0" ::: "memory");
#endif
#endif
}

// ---- fragment loaders -------------------------------------------------------
// A: 16x32 bf16 tile at (row0, k0); lanes 0-15 row=lane K={k0..+7, k0+16..+23},
// lanes 16-31 row=lane-16 K={k0+8..+15, k0+24..+31}.
static __device__ __forceinline__ v16bf load_a_frag(const bf16_t* __restrict__ A,
                                                    long lda, int row0, int k0, int Mmax) {
  int lane = threadIdx.x & 31;
  int hi = lane >> 4;
  int r = row0 + (lane & 15);
  if (r >= Mmax) r = Mmax - 1;
  const bf16_t* p = A + (long)r * lda + k0 + hi * 8;
  v8bf c0 = *(const v8bf*)p;
  v8bf c1 = *(const v8bf*)(p + 16);
  v16bf out;
#pragma unroll
  for (int i = 0; i < 8; ++i) { out[i] = c0[i]; out[i + 8] = c1[i]; }
  return out;
}

// B: 32x16 tile read from an LDS-staged panel laid out [col][KSTEP].
static __device__ __forceinline__ v16bf load_b_lds(const bf16_t* base, int colSub) {
  int lane = threadIdx.x & 31;
  int hi = lane >> 4;
  const bf16_t* p = base + (colSub + (lane & 15)) * KSTEP + hi * 8;
  v8bf c0 = *(const v8bf*)p;
  v8bf c1 = *(const v8bf*)(p + 16);
  v16bf out;
#pragma unroll
  for (int i = 0; i < 8; ++i) { out[i] = c0[i]; out[i + 8] = c1[i]; }
  return out;
}

// B: 32x16 tile directly from pre-transposed global weights [N][K].
static __device__ __forceinline__ v16bf load_b_frag(const bf16_t* __restrict__ Wt,
                                                    long ldb, int col0, int k0) {
  int lane = threadIdx.x & 31;
  int hi = lane >> 4;
  const bf16_t* p = Wt + (long)(col0 + (lane & 15)) * ldb + k0 + hi * 8;
  v8bf c0 = *(const v8bf*)p;
  v8bf c1 = *(const v8bf*)(p + 16);
  v16bf out;
#pragma unroll
  for (int i = 0; i < 8; ++i) { out[i] = c0[i]; out[i + 8] = c1[i]; }
  return out;
}

// conv A fragment: row shifted by (tap-1) along sequence, zeros at the edges.
static __device__ __forceinline__ v16bf load_a_conv(const bf16_t* __restrict__ h,
                                                    int row0, int k0, int tap, int Mtot) {
  int lane = threadIdx.x & 31;
  int hi = lane >> 4;
  int m = row0 + (lane & 15);
  if (m >= Mtot) m = Mtot - 1;
  int s = m & (WS - 1);
  int sp = s + tap - 1;
  v16bf out;
#pragma unroll
  for (int i = 0; i < 16; ++i) out[i] = (bf16_t)0.0f;
  if (sp >= 0 && sp < WS) {
    const bf16_t* p = h + (long)(m + tap - 1) * WD + k0 + hi * 8;
    v8bf c0 = *(const v8bf*)p;
    v8bf c1 = *(const v8bf*)(p + 16);
#pragma unroll
    for (int i = 0; i < 8; ++i) { out[i] = c0[i]; out[i + 8] = c1[i]; }
  }
  return out;
}

// ---- prep kernels -----------------------------------------------------------
__global__ __launch_bounds__(256) void k_transpose_cvt(const float* __restrict__ src,
                                                       bf16_t* __restrict__ dst,
                                                       int K, int N) {
  __shared__ float tile[32][33];
  int kb = blockIdx.y * 32, nb = blockIdx.x * 32;
  int tx = threadIdx.x & 31, ty = threadIdx.x >> 5;
#pragma unroll
  for (int i = 0; i < 32; i += 8) {
    int k = kb + ty + i, n = nb + tx;
    tile[ty + i][tx] = (k < K && n < N) ? src[(long)k * N + n] : 0.0f;
  }
  __syncthreads();
#pragma unroll
  for (int i = 0; i < 32; i += 8) {
    int n = nb + ty + i, k = kb + tx;
    if (n < N && k < K) dst[(long)n * K + k] = f2bf(tile[tx][ty + i]);
  }
}

__global__ __launch_bounds__(256) void k_embed(const int* __restrict__ x,
                                               const float* __restrict__ emb,
                                               const float* __restrict__ pos,
                                               bf16_t* __restrict__ h) {
  long m = blockIdx.x;
  int s = (int)(m & (WS - 1));
  int tok = x[m];
  const float* e = emb + (long)tok * WD;
  const float* p = pos + (long)s * WD;
  bf16_t* o = h + m * WD;
  for (int d = threadIdx.x; d < WD; d += 256) o[d] = f2bf(e[d] + p[d]);
}

// ---- conv: hc = sum_t h(s+t-1) @ Wt + b  (block 128x128, staged B) ---------
__global__ __launch_bounds__(256) void k_conv(const bf16_t* __restrict__ h,
                                              const bf16_t* __restrict__ wT,
                                              const float* __restrict__ bias,
                                              bf16_t* __restrict__ hc) {
  __shared__ bf16_t sB[2][BN * KSTEP];          // 2 x 8KB, double buffered
  const int Mtot = WB * WS;
  int w = threadIdx.x >> 5;
  int mg = w & 3, ng = w >> 2;                  // 4 M-groups x 2 N-groups
  int row0 = blockIdx.x * 128 + mg * 32;
  int colB = blockIdx.y * BN;
  int colW = ng * 64;
  v8f acc[2][4] = {};

  auto stage = [&](int buf, int step) {
    int t = step >> 5;                          // 32 k-steps per tap
    int k0 = (step & 31) * KSTEP;
    const bf16_t* wsrc = wT + (long)t * WD * WD;
    for (int c = threadIdx.x; c < (BN * KSTEP) / 8; c += 256) {
      int col = c >> 2, seg = c & 3;
      cp_b128(wsrc + (long)(colB + col) * WD + k0 + seg * 8,
              &sB[buf][col * KSTEP + seg * 8]);
    }
  };

  const int NS = 3 * (WD / KSTEP);              // 96
  stage(0, 0);
  for (int s = 0; s < NS; ++s) {
    int buf = s & 1;
    wait_async();
    __syncthreads();
    if (s + 1 < NS) stage(buf ^ 1, s + 1);
    int t = s >> 5;
    int k0 = (s & 31) * KSTEP;
    v16bf a0 = load_a_conv(h, row0,      k0, t, Mtot);
    v16bf a1 = load_a_conv(h, row0 + 16, k0, t, Mtot);
#pragma unroll
    for (int ni = 0; ni < 4; ++ni) {
      v16bf b = load_b_lds(&sB[buf][0], colW + ni * 16);
      acc[0][ni] = wmma_bf16(a0, b, acc[0][ni]);
      acc[1][ni] = wmma_bf16(a1, b, acc[1][ni]);
    }
    __syncthreads();
  }

  int lane = threadIdx.x & 31, hi = lane >> 4;
#pragma unroll
  for (int mi = 0; mi < 2; ++mi)
#pragma unroll
    for (int ni = 0; ni < 4; ++ni) {
      int n = colB + colW + ni * 16 + (lane & 15);
      float bn = bias[n];
#pragma unroll
      for (int r = 0; r < 8; ++r) {
        int row = row0 + mi * 16 + r + hi * 8;
        hc[(long)row * WD + n] = f2bf(acc[mi][ni][r] + bn);
      }
    }
}

// ---- gate: nodes = hc * sigmoid(hc @ wg1 + bg1) ----------------------------
__global__ __launch_bounds__(256) void k_gate(const bf16_t* __restrict__ hc,
                                              const bf16_t* __restrict__ wT,
                                              const float* __restrict__ bias,
                                              bf16_t* __restrict__ nodes) {
  __shared__ bf16_t sB[2][BN * KSTEP];
  const int Mtot = WB * WS;
  int w = threadIdx.x >> 5;
  int mg = w & 3, ng = w >> 2;
  int row0 = blockIdx.x * 128 + mg * 32;
  int colB = blockIdx.y * BN;
  int colW = ng * 64;
  v8f acc[2][4] = {};

  auto stage = [&](int buf, int step) {
    int k0 = step * KSTEP;
    for (int c = threadIdx.x; c < (BN * KSTEP) / 8; c += 256) {
      int col = c >> 2, seg = c & 3;
      cp_b128(wT + (long)(colB + col) * WD + k0 + seg * 8,
              &sB[buf][col * KSTEP + seg * 8]);
    }
  };

  int lane = threadIdx.x & 31, hi = lane >> 4;
  const bf16_t* pf = hc + (long)(row0 + (lane & 15)) * WD;  // prefetch stream

  const int NS = WD / KSTEP;                    // 32
  stage(0, 0);
  for (int s = 0; s < NS; ++s) {
    int buf = s & 1;
    wait_async();
    __syncthreads();
    if (s + 1 < NS) stage(buf ^ 1, s + 1);
    int k0 = s * KSTEP;
    if (k0 + 2 * KSTEP < WD) __builtin_prefetch(pf + k0 + 2 * KSTEP, 0, 1);
    v16bf a0 = load_a_frag(hc, WD, row0,      k0, Mtot);
    v16bf a1 = load_a_frag(hc, WD, row0 + 16, k0, Mtot);
#pragma unroll
    for (int ni = 0; ni < 4; ++ni) {
      v16bf b = load_b_lds(&sB[buf][0], colW + ni * 16);
      acc[0][ni] = wmma_bf16(a0, b, acc[0][ni]);
      acc[1][ni] = wmma_bf16(a1, b, acc[1][ni]);
    }
    __syncthreads();
  }

#pragma unroll
  for (int mi = 0; mi < 2; ++mi)
#pragma unroll
    for (int ni = 0; ni < 4; ++ni) {
      int n = colB + colW + ni * 16 + (lane & 15);
      float bn = bias[n];
#pragma unroll
      for (int r = 0; r < 8; ++r) {
        int row = row0 + mi * 16 + r + hi * 8;
        long idx = (long)row * WD + n;
        nodes[idx] = f2bf(bf2f(hc[idx]) * sigmoidf_(acc[mi][ni][r] + bn));
      }
    }
}

// ---- level-0 pad: curr[b,i] = nodes[b, min(i, S-2)] ------------------------
__global__ __launch_bounds__(256) void k_pad0(const bf16_t* __restrict__ nodes,
                                              bf16_t* __restrict__ curr) {
  long r = blockIdx.x;
  long b = r >> 11;
  int  i = (int)(r & 2047);
  int  j = i < (WS - 1) ? i : (WS - 2);
  const bf16_t* s = nodes + (b * WS + j) * WD;
  bf16_t* d = curr + r * WD;
  for (int k = threadIdx.x; k < WD; k += 256) d[k] = s[k];
}

// ---- tree merge: fused 3-output GEMM, K=2D, staged B (3x) ------------------
__global__ __launch_bounds__(256) void k_merge_gemm(const bf16_t* __restrict__ comb,
                                                    const bf16_t* __restrict__ wvT,
                                                    const bf16_t* __restrict__ wgT,
                                                    const bf16_t* __restrict__ wrT,
                                                    const float* __restrict__ bv,
                                                    const float* __restrict__ bg,
                                                    const float* __restrict__ br,
                                                    float* __restrict__ valg,
                                                    bf16_t* __restrict__ rg,
                                                    int M) {
  __shared__ bf16_t sB[2][3][BN * KSTEP];       // 48KB
  const int K = 2 * WD;
  int w = threadIdx.x >> 5;
  int mg = w & 1, ng = w >> 1;                  // 2 M-groups x 4 N-groups
  int row0 = blockIdx.x * 64 + mg * 32;
  int colB = blockIdx.y * BN;
  int colW = ng * 32;
  v8f av[2][2] = {}, ag[2][2] = {}, ar[2][2] = {};

  auto stage = [&](int buf, int step) {
    int k0 = step * KSTEP;
#pragma unroll
    for (int ws = 0; ws < 3; ++ws) {
      const bf16_t* src = (ws == 0) ? wvT : ((ws == 1) ? wgT : wrT);
      for (int c = threadIdx.x; c < (BN * KSTEP) / 8; c += 256) {
        int col = c >> 2, seg = c & 3;
        cp_b128(src + (long)(colB + col) * K + k0 + seg * 8,
                &sB[buf][ws][col * KSTEP + seg * 8]);
      }
    }
  };

  int lane = threadIdx.x & 31, hi = lane >> 4;
  int rclamp = row0 + (lane & 15); if (rclamp >= M) rclamp = M - 1;
  const bf16_t* pf = comb + (long)rclamp * K;   // A prefetch stream

  const int NS = K / KSTEP;                     // 64
  stage(0, 0);
  for (int s = 0; s < NS; ++s) {
    int buf = s & 1;
    wait_async();
    __syncthreads();
    if (s + 1 < NS) stage(buf ^ 1, s + 1);
    int k0 = s * KSTEP;
    if (k0 + 2 * KSTEP < K) __builtin_prefetch(pf + k0 + 2 * KSTEP, 0, 1);
    v16bf a0 = load_a_frag(comb, K, row0,      k0, M);
    v16bf a1 = load_a_frag(comb, K, row0 + 16, k0, M);
#pragma unroll
    for (int ni = 0; ni < 2; ++ni) {
      v16bf bvf = load_b_lds(&sB[buf][0][0], colW + ni * 16);
      v16bf bgf = load_b_lds(&sB[buf][1][0], colW + ni * 16);
      v16bf brf = load_b_lds(&sB[buf][2][0], colW + ni * 16);
      av[0][ni] = wmma_bf16(a0, bvf, av[0][ni]);
      av[1][ni] = wmma_bf16(a1, bvf, av[1][ni]);
      ag[0][ni] = wmma_bf16(a0, bgf, ag[0][ni]);
      ag[1][ni] = wmma_bf16(a1, bgf, ag[1][ni]);
      ar[0][ni] = wmma_bf16(a0, brf, ar[0][ni]);
      ar[1][ni] = wmma_bf16(a1, brf, ar[1][ni]);
    }
    __syncthreads();
  }

#pragma unroll
  for (int mi = 0; mi < 2; ++mi)
#pragma unroll
    for (int ni = 0; ni < 2; ++ni) {
      int n = colB + colW + ni * 16 + (lane & 15);
      float bvn = bv[n], bgn = bg[n], brn = br[n];
#pragma unroll
      for (int r = 0; r < 8; ++r) {
        int row = row0 + mi * 16 + r + hi * 8;
        if (row < M) {
          float v = av[mi][ni][r] + bvn;
          float g = sigmoidf_(ag[mi][ni][r] + bgn);
          valg[(long)row * WD + n] = v * g;
          rg[(long)row * WD + n]   = f2bf(sigmoidf_(ar[mi][ni][r] + brn));
        }
      }
    }
}

// ---- rmsnorm + gated residual combine, one row per block -------------------
__global__ __launch_bounds__(256) void k_combine(const float* __restrict__ valg,
                                                 const bf16_t* __restrict__ rg,
                                                 const bf16_t* __restrict__ cin,
                                                 const float* __restrict__ rmsw,
                                                 bf16_t* __restrict__ cout) {
  __shared__ float red[256];
  long row = blockIdx.x;
  const float* vr = valg + row * WD;
  float ss = 0.0f;
  for (int d = threadIdx.x; d < WD; d += 256) { float v = vr[d]; ss += v * v; }
  red[threadIdx.x] = ss;
  __syncthreads();
  for (int off = 128; off > 0; off >>= 1) {
    if (threadIdx.x < off) red[threadIdx.x] += red[threadIdx.x + off];
    __syncthreads();
  }
  float scale = rsqrtf(red[0] * (1.0f / (float)WD) + 1.1920929e-07f);
  const bf16_t* left  = cin + (2 * row) * WD;
  const bf16_t* right = left + WD;
  const bf16_t* gr    = rg + row * WD;
  bf16_t* o = cout + row * WD;
  for (int d = threadIdx.x; d < WD; d += 256) {
    float merged = vr[d] * scale * rmsw[d];
    float g = bf2f(gr[d]);
    float res = 0.5f * (bf2f(left[d]) + bf2f(right[d]));
    o[d] = f2bf(g * merged + (1.0f - g) * res);
  }
}

// ---- context concat: [root(b), nodes(b, S-2)] ------------------------------
__global__ __launch_bounds__(256) void k_context(const bf16_t* __restrict__ root,
                                                 const bf16_t* __restrict__ nodes,
                                                 bf16_t* __restrict__ ctx) {
  long b = blockIdx.x;
  const bf16_t* r = root + b * WD;
  const bf16_t* l = nodes + (b * WS + (WS - 2)) * WD;
  bf16_t* o = ctx + b * 2 * WD;
  for (int d = threadIdx.x; d < WD; d += 256) { o[d] = r[d]; o[WD + d] = l[d]; }
}

// ---- final projection: out[32, V] = ctx @ wp + bp --------------------------
__global__ __launch_bounds__(256) void k_proj(const bf16_t* __restrict__ ctx,
                                              const bf16_t* __restrict__ wpT,
                                              const float* __restrict__ bp,
                                              float* __restrict__ out) {
  const int K = 2 * WD;
  int wave = threadIdx.x >> 5;
  int col0 = blockIdx.x * 128 + wave * 16;
  v8f a0 = {}, a1 = {};
  for (int k0 = 0; k0 < K; k0 += KSTEP) {
    v16bf A0 = load_a_frag(ctx, K, 0,  k0, WB);
    v16bf A1 = load_a_frag(ctx, K, 16, k0, WB);
    v16bf B  = load_b_frag(wpT, K, col0, k0);
    a0 = wmma_bf16(A0, B, a0);
    a1 = wmma_bf16(A1, B, a1);
  }
  int lane = threadIdx.x & 31, hi = lane >> 4, n = col0 + (lane & 15);
  float bn = bp[n];
#pragma unroll
  for (int r = 0; r < 8; ++r) {
    out[(long)(r + hi * 8) * WV + n]      = a0[r] + bn;
    out[(long)(16 + r + hi * 8) * WV + n] = a1[r] + bn;
  }
}

// ---- host launch ------------------------------------------------------------
extern "C" void kernel_launch(void* const* d_in, const int* in_sizes, int n_in,
                              void* d_out, int out_size, void* d_ws, size_t ws_size,
                              hipStream_t stream) {
  (void)in_sizes; (void)n_in; (void)out_size; (void)ws_size;
  const int*   x      = (const int*)  d_in[0];
  const float* emb    = (const float*)d_in[1];
  const float* pos    = (const float*)d_in[2];
  const float* conv_w = (const float*)d_in[3];
  const float* conv_b = (const float*)d_in[4];
  const float* wg1    = (const float*)d_in[5];
  const float* bg1    = (const float*)d_in[6];
  const float* wmv    = (const float*)d_in[7];
  const float* bmv    = (const float*)d_in[8];
  const float* wmg    = (const float*)d_in[9];
  const float* bmg    = (const float*)d_in[10];
  const float* wrg    = (const float*)d_in[11];
  const float* brg    = (const float*)d_in[12];
  const float* rms_w  = (const float*)d_in[13];
  const float* wp     = (const float*)d_in[14];
  const float* bp     = (const float*)d_in[15];
  float* out = (float*)d_out;

  char* ws = (char*)d_ws;
  size_t off = 0;
  auto alloc = [&](size_t bytes) {
    void* p = ws + off;
    off += (bytes + 255) & ~(size_t)255;
    return p;
  };
  bf16_t* wconvT = (bf16_t*)alloc(3ull * WD * WD * 2);
  bf16_t* wg1T   = (bf16_t*)alloc((size_t)WD * WD * 2);
  bf16_t* wmvT   = (bf16_t*)alloc(2ull * WD * WD * 2);
  bf16_t* wmgT   = (bf16_t*)alloc(2ull * WD * WD * 2);
  bf16_t* wrgT   = (bf16_t*)alloc(2ull * WD * WD * 2);
  bf16_t* wpT    = (bf16_t*)alloc((size_t)WV * 2 * WD * 2);
  bf16_t* h      = (bf16_t*)alloc((size_t)WB * WS * WD * 2);
  bf16_t* hconv  = (bf16_t*)alloc((size_t)WB * WS * WD * 2);
  bf16_t* nodes  = (bf16_t*)alloc((size_t)WB * WS * WD * 2);
  bf16_t* treeA  = (bf16_t*)alloc((size_t)WB * 2048 * WD * 2);
  bf16_t* treeB  = (bf16_t*)alloc((size_t)WB * 1024 * WD * 2);
  float*  valg   = (float*) alloc((size_t)WB * 1024 * WD * 4);
  bf16_t* rgbuf  = (bf16_t*)alloc((size_t)WB * 1024 * WD * 2);
  bf16_t* ctx    = (bf16_t*)alloc((size_t)WB * 2 * WD * 2);

  for (int t = 0; t < 3; ++t)
    k_transpose_cvt<<<dim3(WD / 32, WD / 32), 256, 0, stream>>>(
        conv_w + (size_t)t * WD * WD, wconvT + (size_t)t * WD * WD, WD, WD);
  k_transpose_cvt<<<dim3(WD / 32, WD / 32), 256, 0, stream>>>(wg1, wg1T, WD, WD);
  k_transpose_cvt<<<dim3(WD / 32, (2 * WD) / 32), 256, 0, stream>>>(wmv, wmvT, 2 * WD, WD);
  k_transpose_cvt<<<dim3(WD / 32, (2 * WD) / 32), 256, 0, stream>>>(wmg, wmgT, 2 * WD, WD);
  k_transpose_cvt<<<dim3(WD / 32, (2 * WD) / 32), 256, 0, stream>>>(wrg, wrgT, 2 * WD, WD);
  k_transpose_cvt<<<dim3(WV / 32, (2 * WD) / 32), 256, 0, stream>>>(wp, wpT, 2 * WD, WV);

  k_embed<<<WB * WS, 256, 0, stream>>>(x, emb, pos, h);

  dim3 gBig((WB * WS) / 128, WD / BN);
  k_conv<<<gBig, 256, 0, stream>>>(h, wconvT, conv_b, hconv);
  k_gate<<<gBig, 256, 0, stream>>>(hconv, wg1T, bg1, nodes);

  k_pad0<<<WB * 2048, 256, 0, stream>>>(nodes, treeA);
  bf16_t* cin = treeA;
  bf16_t* cout = treeB;
  int Lin = 2048;
  while (Lin > 1) {
    int Lout = Lin / 2;
    int M = WB * Lout;
    dim3 g((M + 63) / 64, WD / BN);
    k_merge_gemm<<<g, 256, 0, stream>>>(cin, wmvT, wmgT, wrgT, bmv, bmg, brg,
                                        valg, rgbuf, M);
    k_combine<<<M, 256, 0, stream>>>(valg, rgbuf, cin, rms_w, cout);
    bf16_t* tmp = cin; cin = cout; cout = tmp;
    Lin = Lout;
  }

  k_context<<<WB, 256, 0, stream>>>(cin, nodes, ctx);
  k_proj<<<WV / 128, 256, 0, stream>>>(ctx, wpT, bp, out);
}
